// LMPO2TypeI_GTN_78125455115107
// MI455X (gfx1250) — matrix operand order repaired and checked by reading, not compiled, not run on hardware
//
#include <hip/hip_runtime.h>

// ---------------------------------------------------------------------------
// LMPO2 ensemble on MI455X (gfx1250, wave32, WMMA).
//
// Weights precontracted into swizzled f16 GEMM-B operands (~6 MB, L2
// resident). Chain evaluated with v_wmma_f32_16x16x32_f16 (f32 accum).
// Middle site: v_new[s,y] = sum_{p,x} x[s,p]*v[s,x]*G[p,x,y]; a K=32 chunk
// has constant p, so A = x[s,p] * v[s, h*32..+32] -- the raw v fragment is
// p-invariant and loaded ONCE per site; per p only a ds_load_u16 scalar +
// v_pk_mul_f16 remain. One (sample-tile, model) per wave (3072 waves) for
// latency hiding; B double-buffered in registers + global_prefetch ahead.
// Partials reduced by a final kernel (deterministic, no float atomics).
// ---------------------------------------------------------------------------

typedef __attribute__((ext_vector_type(16))) _Float16 v16h;
typedef __attribute__((ext_vector_type(8)))  _Float16 v8h;
typedef __attribute__((ext_vector_type(4)))  _Float16 h4;
typedef __attribute__((ext_vector_type(8)))  float    v8f;
typedef __attribute__((ext_vector_type(4)))  float    f32x4;

#define WAVES 4
#define NTILES 512          // 8192 samples / 16

// workspace layout (units: _Float16 elements unless noted)
static constexpr size_t V0OFF = 0;                      // 5 models * 64*64
static constexpr size_t C1OFF = V0OFF + 5 * 4096;       // 64*16 (o padded)
static constexpr size_t GLOFF = C1OFF + 1024;           // 5 * 4096*16
static constexpr size_t GMOFF = GLOFF + 5 * 65536;      // 10 * 4096*64
static constexpr size_t WS_H  = GMOFF + 10 * 262144;    // halves used by f16
// f32 partial region starts at byte offset WS_H*2 (4-byte aligned)

// ---- B-matrix swizzle: lane l holds column N=l%16; lanes 0-15 hold K rows
// kc*32+0..15, lanes 16-31 hold kc*32+16..31; half t within lane = K offset.
__host__ __device__ __forceinline__ size_t swz(int K, int N, int NT) {
  int kc = K >> 5, kr = K & 31;
  int nt = N >> 4, nl = N & 15;
  int lane = ((kr >> 4) << 4) | nl;
  int t = kr & 15;
  return ((size_t)(kc * NT + nt)) * 512 + (size_t)lane * 16 + t;
}

// ---------------------------------------------------------------------------
// Prep kernels: contract raw f32 params -> swizzled f16 GEMM-B operands.
// ---------------------------------------------------------------------------

__global__ void prep_c1(const float* __restrict__ mpo, const float* __restrict__ mps,
                        _Float16* __restrict__ dst) {
  int idx = blockIdx.x * blockDim.x + threadIdx.x;   // 1024
  int p = idx >> 4, o = idx & 15;
  float s = 0.f;
  if (o < 10)
    for (int r = 0; r < 16; ++r) s += mpo[p * 16 + r] * mps[r * 10 + o];
  dst[swz(p, o, 1)] = (_Float16)s;
}

__global__ void prep_v0(const float* __restrict__ mpo, const float* __restrict__ mps,
                        _Float16* __restrict__ dst) {
  int idx = blockIdx.x * blockDim.x + threadIdx.x;   // 4096
  int p = idx >> 6, n = idx & 63;
  int c = n >> 3, d = n & 7;
  float s = 0.f;
  for (int r = 0; r < 16; ++r) s += mpo[(p * 16 + r) * 8 + c] * mps[r * 8 + d];
  dst[swz(p, n, 4)] = (_Float16)s;
}

__global__ void prep_gmid(const float* __restrict__ mpo, const float* __restrict__ mps,
                          _Float16* __restrict__ dst) {
  int idx = blockIdx.x * blockDim.x + threadIdx.x;   // 262144
  int K = idx >> 6, n = idx & 63;
  int p = K >> 6, a = (K >> 3) & 7, b = K & 7;
  int c = n >> 3, d = n & 7;
  float s = 0.f;
  for (int r = 0; r < 16; ++r)
    s += mpo[(((a * 64 + p) * 16) + r) * 8 + c] * mps[(b * 16 + r) * 8 + d];
  dst[swz(K, n, 4)] = (_Float16)s;
}

__global__ void prep_glast(const float* __restrict__ mpo, const float* __restrict__ mps,
                           _Float16* __restrict__ dst) {
  int idx = blockIdx.x * blockDim.x + threadIdx.x;   // 65536
  int K = idx >> 4, o = idx & 15;
  int p = K >> 6, a = (K >> 3) & 7, b = K & 7;
  float s = 0.f;
  if (o < 10)
    for (int r = 0; r < 16; ++r)
      s += mpo[(a * 64 + p) * 16 + r] * mps[(b * 16 + r) * 10 + o];
  dst[swz(K, o, 1)] = (_Float16)s;
}

// ---------------------------------------------------------------------------
// Main kernel helpers
// ---------------------------------------------------------------------------

__device__ __forceinline__ v8f wmma16(v16h a, v16h b, v8f c) {
  return __builtin_amdgcn_wmma_f32_16x16x32_f16(false, a, false, b, (short)0, c,
                                                false, false);
}

// A-side fragment from f16 LDS [16][64]: lane l -> row M=l%16; hi=l/16;
// half t -> K = t + 16*hi. Two 8-half runs -> 2x ds_load_b128.
__device__ __forceinline__ v16h frag_a(const _Float16* vbuf, int lane, int xx0) {
  int M = lane & 15, hi = lane >> 4;
  const _Float16* row = vbuf + M * 64 + xx0 + 16 * hi;
  v8h r1 = *(const v8h*)(row);
  v8h r2 = *(const v8h*)(row + 8);
  v16h A;
#pragma unroll
  for (int i = 0; i < 8; ++i) { A[i] = r1[i]; A[i + 8] = r2[i]; }
  return A;
}

__device__ __forceinline__ v16h scale_a(v16h a, _Float16 s) {
  v16h r;
#pragma unroll
  for (int i = 0; i < 16; ++i) r[i] = a[i] * s;   // -> v_pk_mul_f16
  return r;
}

__device__ __forceinline__ v16h load_b1(const _Float16* g, int tileIdx, int lane) {
  return *(const v16h*)(g + (size_t)tileIdx * 512 + (size_t)lane * 16);
}

__device__ __forceinline__ void load_b4(v16h (&b)[4], const _Float16* g,
                                        int tileBase, int lane) {
#pragma unroll
  for (int i = 0; i < 4; ++i) b[i] = load_b1(g, tileBase + i, lane);
}

__device__ __forceinline__ void pref_tiles(const _Float16* g, int tileBase,
                                           int n, int lane) {
#pragma unroll
  for (int i = 0; i < 8; ++i)
    if (i < n)
      __builtin_prefetch(g + (size_t)(tileBase + i) * 512 + (size_t)lane * 16, 0, 3);
}

// Scatter a 16x16 f32 C fragment into f16 LDS v buffer [16][64] at col nt*16.
__device__ __forceinline__ void store_c(_Float16* vdst, int lane, int nt, v8f c) {
  int hi = lane >> 4, n = nt * 16 + (lane & 15);
#pragma unroll
  for (int k = 0; k < 8; ++k) vdst[(k + 8 * hi) * 64 + n] = (_Float16)c[k];
}

// ---------------------------------------------------------------------------
// Main kernel: one (sample-tile, model) per wave; writes 16x16 f32 partial.
// ---------------------------------------------------------------------------

__global__ __launch_bounds__(WAVES * 32)
void lmpo_main(const float* __restrict__ x, const _Float16* __restrict__ ws,
               float* __restrict__ part) {
  __shared__ _Float16 smem[WAVES * 3 * 1024];  // per wave: xh, vA, vB (f16)
  const int lane = threadIdx.x & 31;
  const int wave = threadIdx.x >> 5;
  _Float16* xh = smem + wave * 3 * 1024;
  _Float16* vA = xh + 1024;
  _Float16* vB = vA + 1024;

  const int w = blockIdx.x * WAVES + wave;     // 0..3071
  const int tile  = w & (NTILES - 1);
  const int model = w >> 9;                    // 0..5 -> L = model+1
  const int s0 = tile * 16;

  // x tile [16,64] -> f16 LDS
  const f32x4* xg = (const f32x4*)(x + (size_t)s0 * 64);
#pragma unroll
  for (int i = 0; i < 8; ++i) {
    f32x4 v = xg[lane + i * 32];
    h4 o;
#pragma unroll
    for (int j = 0; j < 4; ++j) o[j] = (_Float16)v[j];
    *(h4*)(xh + (size_t)(lane + i * 32) * 4) = o;
  }

  v8f oacc = {};

  if (model == 0) {
    // ---- L = 1: out = x @ C1 (K=64)
    const v16h ax0 = frag_a(xh, lane, 0);
    const v16h ax1 = frag_a(xh, lane, 32);
    oacc = wmma16(ax0, load_b1(ws + C1OFF, 0, lane), oacc);
    oacc = wmma16(ax1, load_b1(ws + C1OFF, 1, lane), oacc);
  } else {
    const int L = model + 1;

    // ---- site 0: v = x @ V0  (M=16, N=64, K=64)
    {
      const v16h ax0 = frag_a(xh, lane, 0);
      const v16h ax1 = frag_a(xh, lane, 32);
      const _Float16* v0 = ws + V0OFF + (size_t)(model - 1) * 4096;
      v16h bk0[4], bk1[4];
      load_b4(bk0, v0, 0, lane);
      load_b4(bk1, v0, 4, lane);
#pragma unroll
      for (int nt = 0; nt < 4; ++nt) {
        v8f acc = {};
        acc = wmma16(ax0, bk0[nt], acc);
        acc = wmma16(ax1, bk1[nt], acc);
        store_c(vA, lane, nt, acc);
      }
    }

    _Float16* cur = vA;
    _Float16* nxt = vB;

    // ---- middle sites: v_new = u @ Gmid (K=4096, N=64)
    const int nmid = L - 2;
    const int gmb = (L >= 3) ? ((L - 3) * (L - 2)) / 2 : 0;
#pragma unroll 1
    for (int i = 0; i < nmid; ++i) {
      const _Float16* g = ws + GMOFF + (size_t)(gmb + i) * 262144;
      // p-invariant raw v fragments (loaded once per site)
      const v16h rv0 = frag_a(cur, lane, 0);
      const v16h rv1 = frag_a(cur, lane, 32);
      v8f acc0 = {}, acc1 = {}, acc2 = {}, acc3 = {};
      v16h b0[4], b1[4];
      load_b4(b0, g, 0, lane);                 // (p=0, h=0)
#pragma unroll 1
      for (int p = 0; p < 64; ++p) {
        _Float16 xs = xh[(lane & 15) * 64 + p];
        if (p + 4 < 64) pref_tiles(g, (p + 4) * 8, 8, lane);
        load_b4(b1, g, p * 8 + 4, lane);       // (p, h=1)
        v16h a0 = scale_a(rv0, xs);
        acc0 = wmma16(a0, b0[0], acc0);
        acc1 = wmma16(a0, b0[1], acc1);
        acc2 = wmma16(a0, b0[2], acc2);
        acc3 = wmma16(a0, b0[3], acc3);
        if (p + 1 < 64) load_b4(b0, g, (p + 1) * 8, lane);  // (p+1, h=0)
        v16h a1 = scale_a(rv1, xs);
        acc0 = wmma16(a1, b1[0], acc0);
        acc1 = wmma16(a1, b1[1], acc1);
        acc2 = wmma16(a1, b1[2], acc2);
        acc3 = wmma16(a1, b1[3], acc3);
      }
      store_c(nxt, lane, 0, acc0);
      store_c(nxt, lane, 1, acc1);
      store_c(nxt, lane, 2, acc2);
      store_c(nxt, lane, 3, acc3);
      _Float16* t = cur; cur = nxt; nxt = t;
    }

    // ---- last site: out = u_last @ Glast (K=4096, N=16; cols 10..15 zero)
    {
      const _Float16* gl = ws + GLOFF + (size_t)(model - 1) * 65536;
      const v16h rv0 = frag_a(cur, lane, 0);
      const v16h rv1 = frag_a(cur, lane, 32);
      v16h e0[2], e1[2];
      e0[0] = load_b1(gl, 0, lane);
      e0[1] = load_b1(gl, 1, lane);
#pragma unroll 1
      for (int p = 0; p < 64; p += 2) {
        e1[0] = load_b1(gl, (p + 1) * 2 + 0, lane);
        e1[1] = load_b1(gl, (p + 1) * 2 + 1, lane);
        if (p + 4 < 64) pref_tiles(gl, (p + 4) * 2, 4, lane);
        {
          _Float16 xs = xh[(lane & 15) * 64 + p];
          oacc = wmma16(scale_a(rv0, xs), e0[0], oacc);
          oacc = wmma16(scale_a(rv1, xs), e0[1], oacc);
        }
        if (p + 2 < 64) {
          e0[0] = load_b1(gl, (p + 2) * 2 + 0, lane);
          e0[1] = load_b1(gl, (p + 2) * 2 + 1, lane);
        }
        {
          _Float16 xs = xh[(lane & 15) * 64 + p + 1];
          oacc = wmma16(scale_a(rv0, xs), e1[0], oacc);
          oacc = wmma16(scale_a(rv1, xs), e1[1], oacc);
        }
      }
    }
  }

  // ---- store 16x16 f32 partial tile: part[(model*512+tile)*256 + M*16 + N]
  float* pt = part + ((size_t)(model * NTILES + tile)) * 256;
  const int hi = lane >> 4, n = lane & 15;
#pragma unroll
  for (int k = 0; k < 8; ++k) pt[(k + 8 * hi) * 16 + n] = oacc[k];
}

// ---------------------------------------------------------------------------
// Reduce: out[s,o] = sum_m part[m][s/16][ (s%16)*16 + o ]
// ---------------------------------------------------------------------------
__global__ void reduce_out(const float* __restrict__ part, float* __restrict__ out) {
  int idx = blockIdx.x * blockDim.x + threadIdx.x;
  if (idx >= 8192 * 10) return;
  int s = idx / 10, o = idx - s * 10;
  int t = s >> 4, r = s & 15;
  float acc = 0.f;
#pragma unroll
  for (int m = 0; m < 6; ++m)
    acc += part[((size_t)(m * NTILES + t)) * 256 + r * 16 + o];
  out[idx] = acc;
}

// ---------------------------------------------------------------------------
// Host launcher
// ---------------------------------------------------------------------------
extern "C" void kernel_launch(void* const* d_in, const int* in_sizes, int n_in,
                              void* d_out, int out_size, void* d_ws, size_t ws_size,
                              hipStream_t stream) {
  (void)in_sizes; (void)n_in; (void)out_size; (void)ws_size;
  const float* x = (const float*)d_in[0];
  _Float16* ws = (_Float16*)d_ws;
  float* part = (float*)((char*)d_ws + WS_H * 2);
  float* out = (float*)d_out;

  // pytree flatten order: per L, mpos[0..L-1] then mpss[0..L-1]
  static const int MPO0[5] = {3, 7, 13, 21, 31};
  static const int MPOL[5] = {4, 9, 16, 25, 36};
  static const int MPS0[5] = {5, 10, 17, 26, 37};
  static const int MPSL[5] = {6, 12, 20, 30, 42};
  static const int MIDO[5] = {-1, 8, 14, 22, 32};
  static const int MIDS[5] = {-1, 11, 18, 27, 38};

  prep_c1<<<4, 256, 0, stream>>>((const float*)d_in[1], (const float*)d_in[2],
                                 ws + C1OFF);

  int gm = 0;
  for (int m = 0; m < 5; ++m) {            // models L = 2..6
    const int L = m + 2;
    prep_v0<<<16, 256, 0, stream>>>((const float*)d_in[MPO0[m]],
                                    (const float*)d_in[MPS0[m]],
                                    ws + V0OFF + (size_t)m * 4096);
    prep_glast<<<256, 256, 0, stream>>>((const float*)d_in[MPOL[m]],
                                        (const float*)d_in[MPSL[m]],
                                        ws + GLOFF + (size_t)m * 65536);
    for (int i = 0; i < L - 2; ++i) {
      prep_gmid<<<1024, 256, 0, stream>>>((const float*)d_in[MIDO[m] + i],
                                          (const float*)d_in[MIDS[m] + i],
                                          ws + GMOFF + (size_t)gm * 262144);
      ++gm;
    }
  }

  // 512 tiles x 6 models = 3072 waves; 4 waves per block
  lmpo_main<<<(NTILES * 6) / WAVES, WAVES * 32, 0, stream>>>(x, ws, part);
  reduce_out<<<(8192 * 10 + 255) / 256, 256, 0, stream>>>(part, out);
}
